// MAGNN_nc_mb_layer_20624432955634
// MI455X (gfx1250) — matrix operand order repaired and checked by reading, not compiled, not run on hardware
//
#include <hip/hip_runtime.h>
#include <hip/hip_bf16.h>
#include <math.h>

typedef __attribute__((ext_vector_type(16))) _Float16 v16h;
typedef __attribute__((ext_vector_type(8)))  float    v8f;

constexpr int P  = 3;
constexpr int E  = 250000;
constexpr int L  = 3;
constexpr int D  = 64;
constexpr int D2 = 32;
constexpr int H  = 8;
constexpr int NT = 8192;
constexpr int HD = 512;   // H * D
constexpr int AD = 128;   // attn vec dim
constexpr int OD = 64;    // out dim
constexpr float LRELU = 0.01f;

__device__ __forceinline__ float wave_sum32(float v) {
#pragma unroll
  for (int s = 16; s > 0; s >>= 1) v += __shfl_xor(v, s, 32);
  return v;
}

// ---------------------------------------------------------------------------
// 1. final_r: cumulative unit complex rotations per metapath. [P][L][D2][2]
// ---------------------------------------------------------------------------
__global__ void final_r_kernel(const float* __restrict__ r_vec,
                               float* __restrict__ final_r) {
  const int lane = threadIdx.x;  // 32 threads, one complex pair each
  if (lane >= D2) return;
  float rf[6][2];
#pragma unroll
  for (int t = 0; t < 3; ++t) {
    float re = r_vec[(t * D2 + lane) * 2 + 0];
    float im = r_vec[(t * D2 + lane) * 2 + 1];
    float inv = rsqrtf(re * re + im * im);
    rf[2 * t + 0][0] = re * inv; rf[2 * t + 0][1] =  im * inv;
    rf[2 * t + 1][0] = re * inv; rf[2 * t + 1][1] = -im * inv;
  }
  const int ET[3][2] = {{0, 1}, {2, 3}, {4, 5}};
#pragma unroll
  for (int p = 0; p < P; ++p) {
    float fr[3], fi[3];
    fr[2] = 1.f; fi[2] = 0.f;
#pragma unroll
    for (int i = 1; i >= 0; --i) {
      const float br = rf[ET[p][i]][0], bi = rf[ET[p][i]][1];
      fr[i] = fr[i + 1] * br - fi[i + 1] * bi;
      fi[i] = fr[i + 1] * bi + fi[i + 1] * br;
    }
#pragma unroll
    for (int l = 0; l < L; ++l) {
      final_r[((p * L + l) * D2 + lane) * 2 + 0] = fr[l];
      final_r[((p * L + l) * D2 + lane) * 2 + 1] = fi[l];
    }
  }
}

// ---------------------------------------------------------------------------
// 2. CSR build: count -> scan -> scatter
// ---------------------------------------------------------------------------
__global__ void count_kernel(const int* __restrict__ dst, int* __restrict__ counts) {
  int i = blockIdx.x * blockDim.x + threadIdx.x;
  if (i >= P * E) return;
  int p = i / E;
  atomicAdd(&counts[p * NT + dst[i]], 1);
}

__global__ void scan_kernel(const int* __restrict__ counts,
                            int* __restrict__ offsets, int* __restrict__ pos) {
  const int p = blockIdx.x;
  const int tid = threadIdx.x;  // 256 threads, 32 elems each
  __shared__ int partial[256];
  const int base = p * NT + tid * 32;
  int s = 0;
#pragma unroll
  for (int i = 0; i < 32; ++i) s += counts[base + i];
  partial[tid] = s;
  __syncthreads();
  if (tid == 0) {
    int run = 0;
    for (int i = 0; i < 256; ++i) { int c = partial[i]; partial[i] = run; run += c; }
  }
  __syncthreads();
  int run = partial[tid];
#pragma unroll
  for (int i = 0; i < 32; ++i) {
    int c = counts[base + i];
    offsets[p * (NT + 1) + tid * 32 + i] = run;
    pos[p * NT + tid * 32 + i] = run;
    run += c;
  }
  if (tid == 255) offsets[p * (NT + 1) + NT] = run;  // == E
}

__global__ void scatter_kernel(const int* __restrict__ dst, int* __restrict__ pos,
                               int* __restrict__ eord) {
  int i = blockIdx.x * blockDim.x + threadIdx.x;
  if (i >= P * E) return;
  int p = i / E, e = i - p * E;
  int slot = atomicAdd(&pos[p * NT + dst[i]], 1);
  eord[p * E + slot] = e;
}

// ---------------------------------------------------------------------------
// 3. Per-target aggregation: hidden + leaky logits + edge softmax + weighted
//    sum + ELU. One wave per (metapath, target); lane = complex pair.
// ---------------------------------------------------------------------------
__device__ __forceinline__ void edge_eval(
    const float* __restrict__ feat, const int* __restrict__ idx3,
    const float* fr_re, const float* fr_im,
    const float* at_re, const float* at_im,
    int lane, float& hr, float& hi, float* a) {
  hr = 0.f; hi = 0.f;
#pragma unroll
  for (int l = 0; l < L; ++l) {
    const int node = idx3[l];
    const float2 f = *(const float2*)(feat + (size_t)node * D + 2 * lane);
    hr += f.x * fr_re[l] - f.y * fr_im[l];
    hi += f.x * fr_im[l] + f.y * fr_re[l];
  }
  hr *= (1.f / 3.f);
  hi *= (1.f / 3.f);
#pragma unroll
  for (int h = 0; h < H; ++h) {
    float part = hr * at_re[h] + hi * at_im[h];
    part = wave_sum32(part);
    a[h] = part > 0.f ? part : LRELU * part;
  }
}

__global__ void aggregate_kernel(const float* __restrict__ feat,
                                 const int* __restrict__ indices,
                                 const int* __restrict__ eord,
                                 const int* __restrict__ offsets,
                                 const float* __restrict__ final_r,
                                 const float* __restrict__ attn,
                                 float* __restrict__ outs) {
  const int wid = blockIdx.x * (blockDim.x >> 5) + (threadIdx.x >> 5);
  if (wid >= P * NT) return;
  const int lane = threadIdx.x & 31;
  const int p = wid / NT;
  const int t = wid - p * NT;

  float fr_re[3], fr_im[3];
#pragma unroll
  for (int l = 0; l < L; ++l) {
    fr_re[l] = final_r[((p * L + l) * D2 + lane) * 2 + 0];
    fr_im[l] = final_r[((p * L + l) * D2 + lane) * 2 + 1];
  }
  float at_re[H], at_im[H];
#pragma unroll
  for (int h = 0; h < H; ++h) {
    at_re[h] = attn[(p * H + h) * D + 2 * lane + 0];
    at_im[h] = attn[(p * H + h) * D + 2 * lane + 1];
  }

  const int off0 = offsets[p * (NT + 1) + t];
  const int off1 = offsets[p * (NT + 1) + t + 1];

  // pass 1: per-head max of leaky logits
  float mx[H];
#pragma unroll
  for (int h = 0; h < H; ++h) mx[h] = -3.0e38f;
  for (int ei = off0; ei < off1; ++ei) {
    const int e = eord[p * E + ei];
    const int* idx3 = indices + ((size_t)p * E + e) * L;
    float hr, hi, a[H];
    edge_eval(feat, idx3, fr_re, fr_im, at_re, at_im, lane, hr, hi, a);
#pragma unroll
    for (int h = 0; h < H; ++h) mx[h] = fmaxf(mx[h], a[h]);
  }

  // pass 2: exp-weighted accumulation
  float se[H], ar[H], ai[H];
#pragma unroll
  for (int h = 0; h < H; ++h) { se[h] = 0.f; ar[h] = 0.f; ai[h] = 0.f; }
  for (int ei = off0; ei < off1; ++ei) {
    const int e = eord[p * E + ei];
    const int* idx3 = indices + ((size_t)p * E + e) * L;
    float hr, hi, a[H];
    edge_eval(feat, idx3, fr_re, fr_im, at_re, at_im, lane, hr, hi, a);
#pragma unroll
    for (int h = 0; h < H; ++h) {
      const float w = __expf(a[h] - mx[h]);
      se[h] += w;
      ar[h] += w * hr;
      ai[h] += w * hi;
    }
  }

  float* orow = outs + ((size_t)p * NT + t) * HD;
#pragma unroll
  for (int h = 0; h < H; ++h) {
    const float inv = (off1 > off0) ? (1.f / se[h]) : 0.f;
    float vr = ar[h] * inv, vi = ai[h] * inv;
    vr = vr > 0.f ? vr : (__expf(vr) - 1.f);  // ELU
    vi = vi > 0.f ? vi : (__expf(vi) - 1.f);
    orow[h * D + 2 * lane + 0] = vr;
    orow[h * D + 2 * lane + 1] = vi;
  }
}

// ---------------------------------------------------------------------------
// 4. WMMA GEMM: C[M,N] = A[M,K] (f32, row major) x W[N,K]^T  (+ bias)
//    Each wave computes one 16-row M tile x two adjacent 16-col N tiles,
//    reusing the A fragment across both WMMAs (halves A traffic).
//    mode 0: s = tanh(C + bias); atomicAdd column partial sums (fc1 path)
//    mode 1: store C + bias (h_fc path)
//    A/B fragments both use contiguous k = 16*(lane/16) + s (wave32).
// ---------------------------------------------------------------------------
__global__ void wmma_gemm_kernel(const float* __restrict__ A, size_t a_zstride,
                                 const float* __restrict__ W,
                                 const float* __restrict__ bias,
                                 float* __restrict__ out,
                                 int M, int N, int K, int mode) {
  const int lane = threadIdx.x & 31;
  const int wave = threadIdx.x >> 5;
  const int tile_m = blockIdx.x * 4 + wave;
  const int tile_n = blockIdx.y;      // covers 32 columns (two 16-col frags)
  const int z = blockIdx.z;
  if (tile_m * 16 >= M) return;
  A += (size_t)z * a_zstride;

  const int mrow = lane & 15;         // A: row within 16-row tile
  const int koff = (lane >> 4) * 16;  // A: K sub-offset per lane half
  const int ncol = lane & 15;         // B/C: column within 16-col fragment
  const int kb   = (lane >> 4) * 16;  // B: K half offset
  const int n0 = tile_n * 32 + ncol;  // fragment 0 column
  const int n1 = n0 + 16;             // fragment 1 column

  const float* arow  = A + (size_t)(tile_m * 16 + mrow) * K;
  const float* wrow0 = W + (size_t)n0 * K + kb;
  const float* wrow1 = W + (size_t)n1 * K + kb;

  v8f acc0 = {}, acc1 = {};
  for (int k0 = 0; k0 < K; k0 += 32) {
    v16h av, bv0, bv1;
#pragma unroll
    for (int i = 0; i < 8; ++i) av[i] = (_Float16)arow[k0 + koff + i];
#pragma unroll
    for (int i = 0; i < 8; ++i) av[8 + i] = (_Float16)arow[k0 + koff + 8 + i];
#pragma unroll
    for (int i = 0; i < 16; ++i) bv0[i] = (_Float16)wrow0[k0 + i];
#pragma unroll
    for (int i = 0; i < 16; ++i) bv1[i] = (_Float16)wrow1[k0 + i];
    acc0 = __builtin_amdgcn_wmma_f32_16x16x32_f16(
        false, av, false, bv0, (short)0, acc0, false, false);
    acc1 = __builtin_amdgcn_wmma_f32_16x16x32_f16(
        false, av, false, bv1, (short)0, acc1, false, false);
  }

  const float b0 = bias[n0];
  const float b1 = bias[n1];
  if (mode == 0) {
    float* colsum = out + z * AD;
    float s0 = 0.f, s1 = 0.f;
#pragma unroll
    for (int r = 0; r < 8; ++r) { s0 += tanhf(acc0[r] + b0); s1 += tanhf(acc1[r] + b1); }
    s0 += __shfl_xor(s0, 16, 32);       // fold M=0..7 with M=8..15 lanes
    s1 += __shfl_xor(s1, 16, 32);
    if (lane < 16) {
      atomicAdd(&colsum[n0], s0);
      atomicAdd(&colsum[n1], s1);
    }
  } else {
    const int part = lane >> 4;
#pragma unroll
    for (int r = 0; r < 8; ++r) {
      const int m = tile_m * 16 + part * 8 + r;  // C VGPR r -> M = r + 8*part
      out[(size_t)m * N + n0] = acc0[r] + b0;
      out[(size_t)m * N + n1] = acc1[r] + b1;
    }
  }
}

// ---------------------------------------------------------------------------
// 5. beta = softmax_p( mean_n(tanh(fc1)) . fc2 )
// ---------------------------------------------------------------------------
__global__ void beta_kernel(const float* __restrict__ colsum,
                            const float* __restrict__ fc2_w,
                            float* __restrict__ beta) {
  const int lane = threadIdx.x;
  float s[P];
#pragma unroll
  for (int p = 0; p < P; ++p) {
    float part = 0.f;
    for (int a = lane; a < AD; a += 32) part += colsum[p * AD + a] * fc2_w[a];
    s[p] = wave_sum32(part) * (1.f / (float)NT);
  }
  const float m = fmaxf(s[0], fmaxf(s[1], s[2]));
  const float e0 = __expf(s[0] - m), e1 = __expf(s[1] - m), e2 = __expf(s[2] - m);
  const float inv = 1.f / (e0 + e1 + e2);
  if (lane == 0) { beta[0] = e0 * inv; beta[1] = e1 * inv; beta[2] = e2 * inv; }
}

// ---------------------------------------------------------------------------
// 6. h[n,d] = sum_p beta_p * outs[p][n][d]   (float4 streaming)
// ---------------------------------------------------------------------------
__global__ void combine_h_kernel(const float* __restrict__ outs,
                                 const float* __restrict__ beta,
                                 float* __restrict__ h) {
  constexpr int N4 = NT * HD / 4;
  const int i = blockIdx.x * blockDim.x + threadIdx.x;
  if (i >= N4) return;
  const float4* o = (const float4*)outs;
  const float b0 = beta[0], b1 = beta[1], b2 = beta[2];
  const float4 a0 = o[i];
  const float4 a1 = o[(size_t)N4 + i];
  const float4 a2 = o[2 * (size_t)N4 + i];
  float4 r;
  r.x = b0 * a0.x + b1 * a1.x + b2 * a2.x;
  r.y = b0 * a0.y + b1 * a1.y + b2 * a2.y;
  r.z = b0 * a0.z + b1 * a1.z + b2 * a2.z;
  r.w = b0 * a0.w + b1 * a1.w + b2 * a2.w;
  ((float4*)h)[i] = r;
}

// ---------------------------------------------------------------------------
extern "C" void kernel_launch(void* const* d_in, const int* in_sizes, int n_in,
                              void* d_out, int out_size, void* d_ws, size_t ws_size,
                              hipStream_t stream) {
  const float* features = (const float*)d_in[0];
  const int*   indices  = (const int*)d_in[1];
  const int*   dst      = (const int*)d_in[2];
  const float* r_vec    = (const float*)d_in[3];
  const float* attn     = (const float*)d_in[4];
  const float* fc1_w    = (const float*)d_in[5];
  const float* fc1_b    = (const float*)d_in[6];
  const float* fc2_w    = (const float*)d_in[7];
  const float* fc_w     = (const float*)d_in[8];
  const float* fc_b     = (const float*)d_in[9];

  float* out  = (float*)d_out;
  float* h_fc = out;                         // [NT, OD]
  float* h    = out + (size_t)NT * OD;       // [NT, HD]

  // workspace layout (~54 MB)
  float* outs   = (float*)d_ws;                // [P, NT, HD] f32 (elu'd ft)
  float* finalr = outs + (size_t)P * NT * HD;  // [P, L, D2, 2]
  float* colsum = finalr + P * L * D2 * 2;     // [P, AD]
  float* betap  = colsum + P * AD;             // [4]
  int* counts   = (int*)(betap + 4);           // [P, NT]
  int* offsets  = counts + P * NT;             // [P, NT+1]
  int* pos      = offsets + P * (NT + 1);      // [P, NT]
  int* eord     = pos + P * NT;                // [P, E]

  hipMemsetAsync(counts, 0, (size_t)P * NT * sizeof(int), stream);
  hipMemsetAsync(colsum, 0, (size_t)P * AD * sizeof(float), stream);

  final_r_kernel<<<1, 32, 0, stream>>>(r_vec, finalr);
  count_kernel<<<(P * E + 255) / 256, 256, 0, stream>>>(dst, counts);
  scan_kernel<<<P, 256, 0, stream>>>(counts, offsets, pos);
  scatter_kernel<<<(P * E + 255) / 256, 256, 0, stream>>>(dst, pos, eord);
  aggregate_kernel<<<P * NT / 8, 256, 0, stream>>>(features, indices, eord,
                                                   offsets, finalr, attn, outs);
  // fc1 GEMM + tanh + column mean partials (per metapath on grid.z)
  wmma_gemm_kernel<<<dim3(NT / 64, AD / 32, P), 128, 0, stream>>>(
      outs, (size_t)NT * HD, fc1_w, fc1_b, colsum, NT, AD, HD, /*mode=*/0);
  beta_kernel<<<1, 32, 0, stream>>>(colsum, fc2_w, betap);
  combine_h_kernel<<<(NT * HD / 4 + 255) / 256, 256, 0, stream>>>(outs, betap, h);
  // h_fc = h @ fc_w^T + fc_b
  wmma_gemm_kernel<<<dim3(NT / 64, OD / 32, 1), 128, 0, stream>>>(
      h, 0, fc_w, fc_b, h_fc, NT, OD, HD, /*mode=*/1);
}
